// FourierAttention_33208687132774
// MI455X (gfx1250) — matrix operand order repaired and checked by reading, not compiled, not run on hardware
//
#include <hip/hip_runtime.h>
#include <hip/hip_bf16.h>
#include <math.h>

// FourierAttention on MI455X (gfx1250), wave32 + WMMA bf16.
//
// Algebraic restructuring (exact):
//   * irfft(A @ vf) == A @ v  (A real)  -> V path is a plain GEMM, no FFT.
//   * sum_{d=0..32} qf_d conj(kf_d) == q^T G k,  G = W^T conj(W) (64x64 complex
//     Gram of the ortho rfft basis)     -> only Q is transformed.
// scores per (b,h):  Re = Re(Qg) @ K^T, Im = Im(Qg) @ K^T  (M=128,N=128,K=4096)
// out    per (b,h):  Out = softmax(|scores|/512) @ V        (M=128,N=4096,K=128)
//
// Qg (Re/Im) and K are stored pre-swizzled in WMMA fragment order so the hot
// GEMM loop is pure global_load_b128 + v_wmma (no LDS staging, no barriers).
// Workspace requirement: ~208 MB.

#define B_ 4
#define L_ 8192
#define H_ 16
#define D_ 64
#define LS_ 64
#define M_ 128         // L/LS
#define BH_ 64         // B*H
#define KDIM_ 4096     // LS*D

typedef __attribute__((ext_vector_type(16))) __bf16 v16bf;
typedef __attribute__((ext_vector_type(8)))  float  v8f;

// ---- wave32 WMMA bf16 fragment layouts (CDNA5 ISA 7.12.2) ----
// A (16x32, 16-bit): lane = mr + 16*hi, hi = bit3 of k_local,
//                    elem = (k_local&7) + 8*(k_local>>4)
// B (32x16, 16-bit): lane = nr + 16*hi, hi = bit3 of k_local,
//                    elem = (k_local&7) + 8*(k_local>>4)
__device__ __forceinline__ int a_koff(int e, bool hi) {
  int p = e >> 1;
  int base = (p < 4) ? (2 * p) : (16 + 2 * (p - 4));
  return base + (e & 1) + (hi ? 8 : 0);
}
__device__ __forceinline__ v16bf load_b_frag_lds(const __bf16* rowT, bool hi) {
  v16bf f;
  int base = hi ? 8 : 0;
#pragma unroll
  for (int e = 0; e < 16; ++e) f[e] = rowT[base + e + (e & 8)];
  return f;
}

// ---------------- kernel 1: G = W^T conj(W), 64x64 complex --------------------
__global__ void fa_init_g(float* __restrict__ Gre, float* __restrict__ Gim) {
  int t = blockIdx.x;   // 0..63
  int tp = threadIdx.x; // 0..63
  float s = (float)(t - tp);
  float re = 0.f, im = 0.f;
  for (int d = 0; d <= 32; ++d) {
    float ang = -6.28318530717958647692f * (float)d * s * (1.0f / 64.0f);
    re += cosf(ang);
    im += sinf(ang);
  }
  Gre[t * 64 + tp] = re * (1.0f / 64.0f);
  Gim[t * 64 + tp] = im * (1.0f / 64.0f);
}

// ------- kernel 2: Qg = q . G (complex) -> bf16, A-fragment-swizzled ----------
// grid.x = BH*8 (bh,mtile), grid.y = 128 (kstep); block = 512 (16 m x 32 k)
// Output: Qsw[(((bh*8+mt)*128 + ks)*32 + lane)*16 + elem]
__global__ void __launch_bounds__(512) fa_transform_q(
    const float* __restrict__ q, const float* __restrict__ Gre,
    const float* __restrict__ Gim, __bf16* __restrict__ QswRe,
    __bf16* __restrict__ QswIm) {
  __shared__ float gre_s[64 * 32];
  __shared__ float gim_s[64 * 32];
  __shared__ float qs[16][64];
  __shared__ __bf16 swRe[32][16];
  __shared__ __bf16 swIm[32][16];
  int bh = blockIdx.x >> 3, mt = blockIdx.x & 7, ks = blockIdx.y;
  int b = bh >> 4, h = bh & 15;
  int k0 = ks * 32;
  int l = k0 >> 6;        // segment time index for this k-chunk
  int tp0 = k0 & 63;      // 0 or 32: which half of the G columns
  int m0 = mt * 16;
  int tid = threadIdx.x;
  for (int i = tid; i < 2048; i += 512) {
    int t = i >> 5, c = i & 31;
    gre_s[i] = Gre[t * 64 + tp0 + c];
    gim_s[i] = Gim[t * 64 + tp0 + c];
  }
  for (int i = tid; i < 1024; i += 512) {
    int mr = i >> 6, t = i & 63;
    qs[mr][t] =
        q[(((size_t)b * L_ + (size_t)(m0 + mr) * LS_ + l) * H_ + h) * D_ + t];
  }
  __syncthreads();
  int mr = tid >> 5, kc = tid & 31; // this thread's (m-row, k_local)
  float re = 0.f, im = 0.f;
#pragma unroll 8
  for (int t = 0; t < 64; ++t) {
    float qv = qs[mr][t];
    re += qv * gre_s[t * 32 + kc];
    im += qv * gim_s[t * 32 + kc];
  }
  int hi = (kc >> 3) & 1;
  int lane = mr + 16 * hi;
  int e = (kc & 7) + 8 * (kc >> 4);
  swRe[lane][e] = (__bf16)re;
  swIm[lane][e] = (__bf16)im;
  __syncthreads();
  if (tid < 64) {
    int lw = tid >> 1, half = tid & 1;
    size_t o = (((size_t)blockIdx.x * 128 + ks) * 32 + lw) * 16 + half * 8;
    *(uint4*)(QswRe + o) = *(const uint4*)(&swRe[lw][half * 8]);
    *(uint4*)(QswIm + o) = *(const uint4*)(&swIm[lw][half * 8]);
  }
}

// ------- kernel 3: K f32 -> bf16, B-fragment-swizzled -------------------------
// grid.x = BH*8 (bh,ntile), grid.y = 128 (kstep); block = 512 (16 n x 32 k)
// Output: Ksw[(((bh*8+nt)*128 + ks)*32 + lane)*16 + elem]
__global__ void __launch_bounds__(512) fa_swizzle_k(
    const float* __restrict__ keys, __bf16* __restrict__ Ksw) {
  __shared__ __bf16 sw[32][16];
  int bh = blockIdx.x >> 3, nt = blockIdx.x & 7, ks = blockIdx.y;
  int b = bh >> 4, h = bh & 15;
  int k0 = ks * 32;
  int l = k0 >> 6, tp0 = k0 & 63;
  int n0 = nt * 16;
  int tid = threadIdx.x;
  int nr = tid >> 5, kc = tid & 31;
  float kv = keys[(((size_t)b * L_ + (size_t)(n0 + nr) * LS_ + l) * H_ + h) * D_ +
                  tp0 + kc];
  int hi = (kc >> 3) & 1;
  int lane = nr + 16 * hi;
  int e = (kc & 7) + 8 * (kc >> 4);
  sw[lane][e] = (__bf16)kv;
  __syncthreads();
  if (tid < 64) {
    int lw = tid >> 1, half = tid & 1;
    size_t o = (((size_t)blockIdx.x * 128 + ks) * 32 + lw) * 16 + half * 8;
    *(uint4*)(Ksw + o) = *(const uint4*)(&sw[lw][half * 8]);
  }
}

// ---------------- kernel 4: scores -> |.|/512 logits (WMMA) -------------------
// grid = (BH, 8 mtiles, 8 ntiles), block = 32.  Pure load_b128 + wmma loop.
__global__ void __launch_bounds__(32) fa_scores(
    const __bf16* __restrict__ QswRe, const __bf16* __restrict__ QswIm,
    const __bf16* __restrict__ Ksw, float* __restrict__ logits) {
  int bh = blockIdx.x, mt = blockIdx.y, nt = blockIdx.z;
  int lane = threadIdx.x;
  v8f accRe = {0, 0, 0, 0, 0, 0, 0, 0};
  v8f accIm = {0, 0, 0, 0, 0, 0, 0, 0};
  const __bf16* pA = QswRe + (((size_t)(bh * 8 + mt) * 128) * 32 + lane) * 16;
  const __bf16* pAi = QswIm + (((size_t)(bh * 8 + mt) * 128) * 32 + lane) * 16;
  const __bf16* pB = Ksw + (((size_t)(bh * 8 + nt) * 128) * 32 + lane) * 16;
#pragma unroll 2
  for (int ks = 0; ks < KDIM_ / 32; ++ks) {
    v16bf aRe = *(const v16bf*)(pA);
    v16bf aIm = *(const v16bf*)(pAi);
    v16bf bF = *(const v16bf*)(pB);
    pA += 32 * 16;
    pAi += 32 * 16;
    pB += 32 * 16;
    __builtin_prefetch(pA, 0, 1);
    __builtin_prefetch(pB, 0, 1);
    accRe = __builtin_amdgcn_wmma_f32_16x16x32_bf16(false, aRe, false, bF,
                                                    (short)0, accRe, false, false);
    accIm = __builtin_amdgcn_wmma_f32_16x16x32_bf16(false, aIm, false, bF,
                                                    (short)0, accIm, false, false);
  }
  // C/D layout: VGPR j -> M = m0+j (+8 for lanes 16..31), N = n0+(lane&15)
#pragma unroll
  for (int j = 0; j < 8; ++j) {
    int m = mt * 16 + j + (lane < 16 ? 0 : 8);
    int n = nt * 16 + (lane & 15);
    float re = accRe[j], im = accIm[j];
    // reference: scores/sqrt(D)=/8 then *1/sqrt(Ls*D)=/64  ->  /512
    logits[((size_t)bh * M_ + m) * M_ + n] =
        sqrtf(re * re + im * im) * (1.0f / 512.0f);
  }
}

// ---------------- kernel 5: row softmax -> bf16 A -----------------------------
__global__ void __launch_bounds__(256) fa_softmax(
    const float* __restrict__ logits, __bf16* __restrict__ A) {
  int row = blockIdx.x * 8 + (threadIdx.x >> 5);
  int lane = threadIdx.x & 31;
  float v[4];
#pragma unroll
  for (int i = 0; i < 4; ++i) v[i] = logits[(size_t)row * M_ + lane + 32 * i];
  float mx = fmaxf(fmaxf(v[0], v[1]), fmaxf(v[2], v[3]));
#pragma unroll
  for (int off = 16; off > 0; off >>= 1) mx = fmaxf(mx, __shfl_xor(mx, off, 32));
  float e[4], s = 0.f;
#pragma unroll
  for (int i = 0; i < 4; ++i) {
    e[i] = __expf(v[i] - mx);
    s += e[i];
  }
#pragma unroll
  for (int off = 16; off > 0; off >>= 1) s += __shfl_xor(s, off, 32);
  float inv = 1.0f / s;
#pragma unroll
  for (int i = 0; i < 4; ++i)
    A[(size_t)row * M_ + lane + 32 * i] = (__bf16)(e[i] * inv);
}

// ---------------- kernel 6: Out = A @ V (WMMA), f32 output --------------------
// grid = (BH, 8 mtiles, 256 ctiles), block = 32. K=128 only -> LDS staging OK.
__global__ void __launch_bounds__(32) fa_out(
    const __bf16* __restrict__ A, const float* __restrict__ values,
    float* __restrict__ out) {
  __shared__ __bf16 Asf[16][132]; // full 16x128 A tile
  __shared__ __bf16 BsT[16][40];  // BsT[c][n]
  int bh = blockIdx.x, mt = blockIdx.y, nt = blockIdx.z;
  int b = bh >> 4, h = bh & 15;
  int lane = threadIdx.x;
  int m0 = mt * 16;
  int c0 = nt * 16;
  int l = c0 >> 6, d0 = c0 & 63; // 16-col tile lies within one l
#pragma unroll 4
  for (int i = 0; i < 64; ++i) {
    int idx = i * 32 + lane;
    int r = idx >> 7, c = idx & 127;
    Asf[r][c] = A[((size_t)bh * M_ + m0 + r) * M_ + c];
  }
  __syncthreads();
  v8f acc = {0, 0, 0, 0, 0, 0, 0, 0};
  for (int ks = 0; ks < 4; ++ks) { // K = 128 = 4 * 32 (reduction over n)
#pragma unroll
    for (int i = 0; i < 16; ++i) {
      int idx = i * 32 + lane;
      int kk = idx >> 4, cc = idx & 15;
      int n = ks * 32 + kk;
      float vv =
          values[(((size_t)b * L_ + (size_t)n * LS_ + l) * H_ + h) * D_ + d0 + cc];
      BsT[cc][kk] = (__bf16)vv;
    }
    __syncthreads();
    bool hi = lane >= 16;
    int mr = lane & 15;
    v16bf aF;
#pragma unroll
    for (int e = 0; e < 16; ++e) aF[e] = Asf[mr][ks * 32 + a_koff(e, hi)];
    v16bf bF = load_b_frag_lds(&BsT[mr][0], hi);
    acc = __builtin_amdgcn_wmma_f32_16x16x32_bf16(false, aF, false, bF,
                                                  (short)0, acc, false, false);
    __syncthreads();
  }
#pragma unroll
  for (int j = 0; j < 8; ++j) {
    int m = m0 + j + (lane < 16 ? 0 : 8);
    int cc = lane & 15;
    out[(((size_t)b * L_ + (size_t)m * LS_ + l) * H_ + h) * D_ + d0 + cc] = acc[j];
  }
}

extern "C" void kernel_launch(void* const* d_in, const int* in_sizes, int n_in,
                              void* d_out, int out_size, void* d_ws,
                              size_t ws_size, hipStream_t stream) {
  const float* queries = (const float*)d_in[0];
  const float* keys = (const float*)d_in[1];
  const float* values = (const float*)d_in[2];
  float* out = (float*)d_out;

  // workspace layout (bytes); total ~208 MB
  char* ws = (char*)d_ws;
  float* Gre = (float*)(ws);
  float* Gim = (float*)(ws + 16384);
  size_t seg = (size_t)BH_ * M_ * KDIM_ * 2; // 67,108,864 B per bf16 tensor
  __bf16* QswRe = (__bf16*)(ws + 32768);
  __bf16* QswIm = (__bf16*)(ws + 32768 + seg);
  __bf16* Ksw = (__bf16*)(ws + 32768 + 2 * seg);
  float* logits = (float*)(ws + 32768 + 3 * seg);
  __bf16* Amat = (__bf16*)(ws + 32768 + 3 * seg + (size_t)BH_ * M_ * M_ * 4);

  fa_init_g<<<dim3(64), dim3(64), 0, stream>>>(Gre, Gim);
  fa_transform_q<<<dim3(BH_ * 8, 128), dim3(512), 0, stream>>>(queries, Gre, Gim,
                                                               QswRe, QswIm);
  fa_swizzle_k<<<dim3(BH_ * 8, 128), dim3(512), 0, stream>>>(keys, Ksw);
  fa_scores<<<dim3(BH_, M_ / 16, M_ / 16), dim3(32), 0, stream>>>(QswRe, QswIm,
                                                                  Ksw, logits);
  fa_softmax<<<dim3(BH_ * M_ / 8), dim3(256), 0, stream>>>(logits, Amat);
  fa_out<<<dim3(BH_, M_ / 16, KDIM_ / 16), dim3(32), 0, stream>>>(Amat, values,
                                                                  out);
}
